// SharedProjectionAssociativeLM_34368328303149
// MI455X (gfx1250) — matrix operand order repaired and checked by reading, not compiled, not run on hardware
//
#include <hip/hip_runtime.h>
#include <cstdint>

typedef __bf16 bf16;
typedef __attribute__((ext_vector_type(16))) __bf16 v16bf;
typedef __attribute__((ext_vector_type(8)))  float  v8f;
typedef __attribute__((ext_vector_type(4)))  unsigned int v4u;
typedef __attribute__((ext_vector_type(8)))  int v8i;
typedef __attribute__((ext_vector_type(4)))  int v4i;

constexpr int VV = 32000, EE = 512, HH = 1024, MM = 256, BB = 2, LL = 2048;

union FragU { uint4 q[2]; v16bf v; };

__device__ __forceinline__ v8f wmma_bf16(v16bf a, v16bf b, v8f c) {
  return __builtin_amdgcn_wmma_f32_16x16x32_bf16(false, a, false, b, (short)0, c,
                                                 false, false);
}

#if __has_builtin(__builtin_amdgcn_tensor_load_to_lds)
#define HAVE_TDM 1
#else
#define HAVE_TDM 0
#endif

// ---------------------------------------------------------------- cast f32->bf16
__global__ void cast_f32_bf16_kernel(const float* __restrict__ s,
                                     bf16* __restrict__ d, int n) {
  for (int i = blockIdx.x * blockDim.x + threadIdx.x; i < n;
       i += gridDim.x * blockDim.x)
    d[i] = (bf16)s[i];
}

// ------------------------------------------------------- embedding gather (bf16)
__global__ void gather_kernel(const int* __restrict__ ids,
                              const float* __restrict__ emb,
                              bf16* __restrict__ x, int total) {
  int i = blockIdx.x * blockDim.x + threadIdx.x;
  if (i < total) {
    int row = i >> 9;           // /EE
    int col = i & (EE - 1);
    x[i] = (bf16)emb[(size_t)ids[row] * EE + col];
  }
}

// ------------------------------------------- WMMA GEMM with TDM-staged A tile
// C[M x N] = A[M x KT](bf16,rm) * Bw[N x KT](bf16,rm)^T + bias[N]
// Block: 256 thr = 8 waves as 2(M) x 4(N); wave tile 32x32 (2x2 reg blocking);
// block tile 64 x 128. A block-tile (64 x KT) staged to LDS by the Tensor Data
// Mover once per block; B streams from L2 with prefetch.
template <int KT>
__global__ void __launch_bounds__(256)
gemm_tdm_kernel(const bf16* __restrict__ A, const bf16* __restrict__ Bw,
                const float* __restrict__ bias,
                float* __restrict__ Cf, bf16* __restrict__ Cb,
                int ldc, int Mrows) {
  __shared__ bf16 As[64 * KT];

#if HAVE_TDM
  if (threadIdx.x < 32) {
    const unsigned ldsOff = (unsigned)(uintptr_t)&As[0];
    const uint64_t ga = (uint64_t)(uintptr_t)(A + (size_t)blockIdx.y * 64 * KT);
    // D# group 0: count=1 | lds_addr | global_addr | type=2
    v4u g0 = {1u, ldsOff, (unsigned)ga,
              (unsigned)((ga >> 32) & 0x1FFFFFFu) | 0x80000000u};
    // D# group 1: data_size=2B; tensor_dim0=KT; tensor_dim1=Mrows;
    //             tile_dim0=KT; tile_dim1=64; tensor_dim0_stride=KT
    v8i g1 = {(int)0x00010000,
              (int)(((unsigned)KT & 0xFFFFu) << 16),
              (int)(((unsigned)KT >> 16) | (((unsigned)Mrows & 0xFFFFu) << 16)),
              (int)(((unsigned)Mrows >> 16) | ((unsigned)KT << 16)),
              64,
              KT, 0, 0};
    v4i gz = {0, 0, 0, 0};
#if defined(__clang_major__) && __clang_major__ >= 23
    v8i gz8 = {0, 0, 0, 0, 0, 0, 0, 0};
    __builtin_amdgcn_tensor_load_to_lds(g0, g1, gz, gz, gz8, 0);
#else
    __builtin_amdgcn_tensor_load_to_lds(g0, g1, gz, gz, 0);
#endif
    __builtin_amdgcn_s_wait_tensorcnt(0);
  }
  __syncthreads();
#else
  {
    const uint4* src = (const uint4*)(A + (size_t)blockIdx.y * 64 * KT);
    uint4* dst = (uint4*)As;
    for (int i = threadIdx.x; i < 64 * KT / 8; i += 256) dst[i] = src[i];
    __syncthreads();
  }
#endif

  const int lane = threadIdx.x & 31;
  const int w    = threadIdx.x >> 5;
  const int lm   = (w & 1) * 32;                       // local M base in As
  const int n0   = blockIdx.x * 128 + (w >> 1) * 32;   // global N base

  const bf16* a0p = As + (size_t)(lm + (lane & 15)) * KT + ((lane & 16) ? 8 : 0);
  const bf16* a1p = a0p + (size_t)16 * KT;
  const bf16* b0p = Bw + (size_t)(n0 + (lane & 15)) * KT + ((lane & 16) ? 8 : 0);
  const bf16* b1p = b0p + (size_t)16 * KT;

  v8f acc00 = {}, acc01 = {}, acc10 = {}, acc11 = {};
  for (int k0 = 0; k0 < KT; k0 += 32) {
    __builtin_prefetch(b0p + k0 + 64, 0, 1);
    __builtin_prefetch(b1p + k0 + 64, 0, 1);
    FragU a0, a1, b0, b1;
    a0.q[0] = *(const uint4*)(a0p + k0);
    a0.q[1] = *(const uint4*)(a0p + k0 + 16);
    a1.q[0] = *(const uint4*)(a1p + k0);
    a1.q[1] = *(const uint4*)(a1p + k0 + 16);
    b0.q[0] = *(const uint4*)(b0p + k0);
    b0.q[1] = *(const uint4*)(b0p + k0 + 16);
    b1.q[0] = *(const uint4*)(b1p + k0);
    b1.q[1] = *(const uint4*)(b1p + k0 + 16);
    acc00 = wmma_bf16(a0.v, b0.v, acc00);
    acc01 = wmma_bf16(a0.v, b1.v, acc01);
    acc10 = wmma_bf16(a1.v, b0.v, acc10);
    acc11 = wmma_bf16(a1.v, b1.v, acc11);
  }

  const int m0 = blockIdx.y * 64 + lm;
  const int hi = (lane & 16) ? 8 : 0;
  const int c0 = n0 + (lane & 15);
  const float bv0 = bias[c0], bv1 = bias[c0 + 16];
  if (Cf) {
#pragma unroll
    for (int r = 0; r < 8; ++r) {
      Cf[(size_t)(m0 + hi + r) * ldc + c0]           = acc00[r] + bv0;
      Cf[(size_t)(m0 + hi + r) * ldc + c0 + 16]      = acc01[r] + bv1;
      Cf[(size_t)(m0 + 16 + hi + r) * ldc + c0]      = acc10[r] + bv0;
      Cf[(size_t)(m0 + 16 + hi + r) * ldc + c0 + 16] = acc11[r] + bv1;
    }
  } else {
#pragma unroll
    for (int r = 0; r < 8; ++r) {
      Cb[(size_t)(m0 + hi + r) * ldc + c0]           = (bf16)(acc00[r] + bv0);
      Cb[(size_t)(m0 + hi + r) * ldc + c0 + 16]      = (bf16)(acc01[r] + bv1);
      Cb[(size_t)(m0 + 16 + hi + r) * ldc + c0]      = (bf16)(acc10[r] + bv0);
      Cb[(size_t)(m0 + 16 + hi + r) * ldc + c0 + 16] = (bf16)(acc11[r] + bv1);
    }
  }
}

// --------------------------------------------------------- persistent GRU kernel
// One 32-wave workgroup. Phase 1 is k-outer / 6-N-tile-inner with 6 persistent
// accumulators: each LDS A fragment is loaded once per k-step and consumed by
// 6 WMMAs immediately (no hoisting -> no scratch spills in the serial loop).
__global__ void __launch_bounds__(1024)
gru_kernel(const float* __restrict__ xp, const bf16* __restrict__ Whh,
           const float* __restrict__ b_hh, float* __restrict__ states,
           bf16* __restrict__ states_bf) {
  __shared__ float hcur[2][HH];        // 8 KB  current hidden (f32 master copy)
  __shared__ bf16  hpad[16][HH];       // 32 KB 16-row-padded A tile (rows>=2 zero)
  __shared__ float hh_s[2][3 * HH];    // 24 KB recurrent pre-activations

  const int tid = threadIdx.x;
  for (int i = tid; i < 2 * HH; i += 1024) hcur[i >> 10][i & (HH - 1)] = 0.f;
  for (int i = tid; i < 16 * HH; i += 1024) (&hpad[0][0])[i] = (bf16)0.f;
  __syncthreads();

  const int lane = tid & 31;
  const int w    = tid >> 5;

  const bf16* ap = &hpad[lane & 15][(lane & 16) ? 8 : 0];
  const bf16* bbase =
      Whh + (size_t)(w * 96 + (lane & 15)) * HH + ((lane & 16) ? 8 : 0);

  for (int t = 0; t < LL; ++t) {
    // ---- phase 1: hh = hpad @ Whh^T  (wave owns 6 consecutive 16-col tiles)
    {
      v8f acc[6] = {};
#pragma unroll 2
      for (int k0 = 0; k0 < HH; k0 += 32) {
        FragU a;
        a.q[0] = *(const uint4*)(ap + k0);
        a.q[1] = *(const uint4*)(ap + k0 + 16);
#pragma unroll
        for (int q = 0; q < 6; ++q) {
          const bf16* bp = bbase + (size_t)(q * 16) * HH + k0;
          FragU b;
          b.q[0] = *(const uint4*)(bp);
          b.q[1] = *(const uint4*)(bp + 16);
          acc[q] = wmma_bf16(a.v, b.v, acc[q]);
        }
      }
      if (lane < 16) {               // only rows 0,1 are live (batch = 2)
#pragma unroll
        for (int q = 0; q < 6; ++q) {
          const int n0 = (w * 6 + q) * 16;
          hh_s[0][n0 + lane] = acc[q][0];
          hh_s[1][n0 + lane] = acc[q][1];
        }
      }
    }
    __syncthreads();
    // ---- phase 2: gates + state update
    for (int i = tid; i < 2 * HH; i += 1024) {
      const int b = i >> 10, j = i & (HH - 1);
      const float* xrow = xp + (size_t)(b * LL + t) * (3 * HH);
      const float x_r = xrow[j], x_z = xrow[HH + j], x_n = xrow[2 * HH + j];
      const float h_r = hh_s[b][j] + b_hh[j];
      const float h_z = hh_s[b][HH + j] + b_hh[HH + j];
      const float h_n = hh_s[b][2 * HH + j] + b_hh[2 * HH + j];
      const float rg = 1.f / (1.f + __expf(-(x_r + h_r)));
      const float zg = 1.f / (1.f + __expf(-(x_z + h_z)));
      const float ng = tanhf(x_n + rg * h_n);
      const float hnew = (1.f - zg) * ng + zg * hcur[b][j];
      hcur[b][j] = hnew;
      hpad[b][j] = (bf16)hnew;
      states[(size_t)(b * LL + t) * HH + j]    = hnew;
      states_bf[(size_t)(b * LL + t) * HH + j] = (bf16)hnew;
    }
    __syncthreads();
  }
}

// ----------------------------------------------------------------- gate (sigmoid)
__global__ void __launch_bounds__(256)
gate_kernel(const float* __restrict__ states, const float* __restrict__ Wg,
            const float* __restrict__ bg, const float* __restrict__ mscale,
            float* __restrict__ gate) {
  const int row  = blockIdx.x * 8 + (threadIdx.x >> 5);
  const int lane = threadIdx.x & 31;
  const float* s = states + (size_t)row * HH;
  float p = 0.f;
  for (int j = lane; j < HH; j += 32) p += s[j] * Wg[j];
  for (int o = 16; o; o >>= 1) p += __shfl_xor(p, o, 32);
  if (lane == 0) gate[row] = (1.f / (1.f + __expf(-(p + bg[0])))) * mscale[0];
}

// --------------------------------------- causal attention + fused scatter-add
__global__ void __launch_bounds__(256)
attn_scatter_kernel(const bf16* __restrict__ memv, const float* __restrict__ gate,
                    const int* __restrict__ ids, float* __restrict__ out) {
  __shared__ float sc[16][2052];       // 128.25 KB score strip (320 KB/WGP LDS)
  const int b    = blockIdx.y;
  const int l0   = blockIdx.x * 16;
  const int lane = threadIdx.x & 31;
  const int w    = threadIdx.x >> 5;
  const bf16* mb = memv + (size_t)b * LL * MM;

  const bf16* ap = mb + (size_t)(l0 + (lane & 15)) * MM + ((lane & 16) ? 8 : 0);
  v16bf afr[8];
#pragma unroll
  for (int kc = 0; kc < 8; ++kc) {
    FragU a;
    a.q[0] = *(const uint4*)(ap + kc * 32);
    a.q[1] = *(const uint4*)(ap + kc * 32 + 16);
    afr[kc] = a.v;
  }

  const int ntiles = l0 / 16 + 1;      // key tiles covering m <= l0+15
  for (int kt = w; kt < ntiles; kt += 8) {
    const int m0 = kt * 16;
    const bf16* bp = mb + (size_t)(m0 + (lane & 15)) * MM + ((lane & 16) ? 8 : 0);
    v8f acc = {};
#pragma unroll
    for (int kc = 0; kc < 8; ++kc) {
      FragU bb;
      bb.q[0] = *(const uint4*)(bp + kc * 32);
      bb.q[1] = *(const uint4*)(bp + kc * 32 + 16);
      acc = wmma_bf16(afr[kc], bb.v, acc);
    }
    const int rb = (lane & 16) ? 8 : 0;
    const int c  = m0 + (lane & 15);
#pragma unroll
    for (int r = 0; r < 8; ++r) sc[rb + r][c] = acc[r] * 0.0625f;  // 1/sqrt(256)
  }
  __syncthreads();

  for (int r = w; r < 16; r += 8) {          // wave handles rows r, r+8
    const int l = l0 + r, nv = l;            // valid keys: m in [0, l)
    if (nv == 0) continue;                   // row 0 contributes exactly 0
    float mx = -3.4e38f;
    for (int m = lane; m < nv; m += 32) mx = fmaxf(mx, sc[r][m]);
    for (int o = 16; o; o >>= 1) mx = fmaxf(mx, __shfl_xor(mx, o, 32));
    float sum = 0.f;
    for (int m = lane; m < nv; m += 32) sum += __expf(sc[r][m] - mx);
    for (int o = 16; o; o >>= 1) sum += __shfl_xor(sum, o, 32);
    const float coef = gate[b * LL + l] / sum;   // gate already * memory_scale
    const int* idr = ids + b * LL;
    float* orow = out + (size_t)(b * LL + l) * VV;
    for (int m = lane; m < nv; m += 32)
      atomicAdd(orow + idr[m], __expf(sc[r][m] - mx) * coef);
  }
}

// ================================================================ host launcher
extern "C" void kernel_launch(void* const* d_in, const int* in_sizes, int n_in,
                              void* d_out, int out_size, void* d_ws, size_t ws_size,
                              hipStream_t stream) {
  const int*   ids      = (const int*)d_in[0];
  const float* emb      = (const float*)d_in[1];
  const float* W_ih     = (const float*)d_in[2];
  const float* W_hh     = (const float*)d_in[3];
  const float* b_ih     = (const float*)d_in[4];
  const float* b_hh     = (const float*)d_in[5];
  const float* W_mem    = (const float*)d_in[6];
  const float* b_mem    = (const float*)d_in[7];
  const float* W_gate   = (const float*)d_in[8];
  const float* b_gate   = (const float*)d_in[9];
  const float* W_h2e    = (const float*)d_in[10];
  const float* b_h2e    = (const float*)d_in[11];
  const float* out_bias = (const float*)d_in[12];
  const float* mscale   = (const float*)d_in[13];
  float* out = (float*)d_out;
  (void)in_sizes; (void)n_in; (void)out_size; (void)ws_size;

  // ---- workspace carve (256 B aligned), ~130 MB total
  size_t off = 0;
  auto take = [&](size_t bytes) -> void* {
    void* p = (char*)d_ws + off;
    off += (bytes + 255) & ~(size_t)255;
    return p;
  };
  bf16*  emb_b  = (bf16*)take((size_t)VV * EE * 2);
  bf16*  Wih_b  = (bf16*)take((size_t)3 * HH * EE * 2);
  bf16*  Whh_b  = (bf16*)take((size_t)3 * HH * HH * 2);
  bf16*  Wh2e_b = (bf16*)take((size_t)EE * HH * 2);
  bf16*  Wmem_b = (bf16*)take((size_t)MM * HH * 2);
  bf16*  x_b    = (bf16*)take((size_t)BB * LL * EE * 2);
  float* xp     = (float*)take((size_t)BB * LL * 3 * HH * 4);
  float* st     = (float*)take((size_t)BB * LL * HH * 4);
  bf16*  st_b   = (bf16*)take((size_t)BB * LL * HH * 2);
  bf16*  proj_b = (bf16*)take((size_t)BB * LL * EE * 2);
  bf16*  mem_b  = (bf16*)take((size_t)BB * LL * MM * 2);
  float* gate   = (float*)take((size_t)BB * LL * 4);

  // ---- 1) weight casts to bf16
  auto cast = [&](const float* s, bf16* d, int n) {
    int g = (n + 1023) / 1024; if (g > 4096) g = 4096;
    cast_f32_bf16_kernel<<<g, 1024, 0, stream>>>(s, d, n);
  };
  cast(emb,   emb_b,  VV * EE);
  cast(W_ih,  Wih_b,  3 * HH * EE);
  cast(W_hh,  Whh_b,  3 * HH * HH);
  cast(W_h2e, Wh2e_b, EE * HH);
  cast(W_mem, Wmem_b, MM * HH);

  // ---- 2) embedding gather
  {
    int total = BB * LL * EE;
    gather_kernel<<<(total + 255) / 256, 256, 0, stream>>>(ids, emb, x_b, total);
  }

  // ---- 3) xp = x @ W_ih^T + b_ih   [4096 x 3072], K=512
  gemm_tdm_kernel<512><<<dim3(3 * HH / 128, BB * LL / 64), 256, 0, stream>>>(
      x_b, Wih_b, b_ih, xp, nullptr, 3 * HH, BB * LL);

  // ---- 4) GRU recurrence (persistent workgroup)
  gru_kernel<<<1, 1024, 0, stream>>>(xp, Whh_b, b_hh, st, st_b);

  // ---- 5) proj_b = states @ W_h2e^T + b_h2e  (bf16 out) [4096 x 512], K=1024
  gemm_tdm_kernel<1024><<<dim3(EE / 128, BB * LL / 64), 256, 0, stream>>>(
      st_b, Wh2e_b, b_h2e, nullptr, proj_b, EE, BB * LL);

  // ---- 6) mem_b = states @ W_mem^T + b_mem  (bf16 out) [4096 x 256], K=1024
  gemm_tdm_kernel<1024><<<dim3(MM / 128, BB * LL / 64), 256, 0, stream>>>(
      st_b, Wmem_b, b_mem, nullptr, mem_b, MM, BB * LL);

  // ---- 7) gate = sigmoid(states @ W_gate^T + b_gate) * memory_scale
  gate_kernel<<<BB * LL / 8, 256, 0, stream>>>(st, W_gate, b_gate, mscale, gate);

  // ---- 8) base = proj @ emb^T + output_bias  -> d_out  [4096 x 32000], K=512
  gemm_tdm_kernel<512><<<dim3(VV / 128, BB * LL / 64), 256, 0, stream>>>(
      proj_b, emb_b, out_bias, out, nullptr, VV, BB * LL);

  // ---- 9) causal attention + gated scatter-add into d_out
  attn_scatter_kernel<<<dim3(LL / 16, BB), 256, 0, stream>>>(mem_b, gate, ids, out);
}
